// MultiHeadAttentionBlock_7705171329045
// MI455X (gfx1250) — matrix operand order repaired and checked by reading, not compiled, not run on hardware
//
#include <hip/hip_runtime.h>
#include <hip/hip_fp16.h>
#include <cstdint>

typedef __attribute__((ext_vector_type(16))) _Float16 v16h;
typedef __attribute__((ext_vector_type(8)))  float    v8f;

union Frag16 { v16h h; uint4 u[2]; };

#define B_  2
#define S_  2048
#define D_  1024
#define H_  16
#define DK_ 64

// ---------------------------------------------------------------------------
// GEMM: Y[M,N] = X[M,K] @ W[N,K]^T + bias, M=4096, N=K=1024.
// MODE 0: X fp32, out f16 scattered to [B,H,S,DK]        (Q, K projections)
// MODE 1: X fp32, out f16 scattered to [B*H, DK, S]      (V projection, pre-transposed)
// MODE 2: X f16,  out fp32 row-major [M,N]               (output projection)
// Block: 256 threads = 8 waves, 128x128 tile, K-step 64, f16 WMMA 16x16x32.
// ---------------------------------------------------------------------------
__device__ inline void cvt16(const float4* src, Frag16& dst) {
    _Float16* h = reinterpret_cast<_Float16*>(&dst);
#pragma unroll
    for (int i = 0; i < 4; ++i) {
        float4 f = src[i];
        h[i * 4 + 0] = (_Float16)f.x;
        h[i * 4 + 1] = (_Float16)f.y;
        h[i * 4 + 2] = (_Float16)f.z;
        h[i * 4 + 3] = (_Float16)f.w;
    }
}

template <int MODE>
__global__ __launch_bounds__(256)
void gemm_xwT_kernel(const void* __restrict__ Xin, const float* __restrict__ W,
                     const float* __restrict__ bias, void* __restrict__ outp)
{
    __shared__ __align__(16) _Float16 As[128][72];   // [row][k0..63], 144B stride
    __shared__ __align__(16) _Float16 Bs[128][72];   // [n][k0..63]

    const float*    Xf = (MODE == 2) ? nullptr : (const float*)Xin;
    const _Float16* Xh = (MODE == 2) ? (const _Float16*)Xin : nullptr;

    const int t    = threadIdx.x;
    const int wv   = t >> 5;
    const int lane = t & 31;
    const int half = lane >> 4;       // 0/1
    const int lm   = lane & 15;
    const int wrow = (wv >> 2) * 64;  // 0 or 64
    const int wcol = (wv & 3) * 32;   // 0,32,64,96

    const int blockM = blockIdx.y * 128;
    const int blockN = blockIdx.x * 128;

    const int lrow = t >> 1;          // 0..127
    const int lcol = (t & 1) * 32;    // 0 or 32 (halves)

    v8f acc[4][2] = {};

    for (int kt = 0; kt < D_ / 64; ++kt) {
        const int k0 = kt * 64;
        // ---- A tile (X): 32 halves per thread ----
        {
            Frag16 tmp0, tmp1;
            if constexpr (MODE == 2) {
                const uint4* src = reinterpret_cast<const uint4*>(
                    Xh + (size_t)(blockM + lrow) * D_ + k0 + lcol);
                tmp0.u[0] = src[0]; tmp0.u[1] = src[1];
                tmp1.u[0] = src[2]; tmp1.u[1] = src[3];
            } else {
                const float4* src = reinterpret_cast<const float4*>(
                    Xf + (size_t)(blockM + lrow) * D_ + k0 + lcol);
                cvt16(src, tmp0);
                cvt16(src + 4, tmp1);
            }
            uint4* dst = reinterpret_cast<uint4*>(&As[lrow][lcol]);
            dst[0] = tmp0.u[0]; dst[1] = tmp0.u[1];
            dst[2] = tmp1.u[0]; dst[3] = tmp1.u[1];
        }
        // ---- B tile (W, fp32 -> f16) ----
        {
            Frag16 tmp0, tmp1;
            const float4* src = reinterpret_cast<const float4*>(
                W + (size_t)(blockN + lrow) * D_ + k0 + lcol);
            cvt16(src, tmp0);
            cvt16(src + 4, tmp1);
            uint4* dst = reinterpret_cast<uint4*>(&Bs[lrow][lcol]);
            dst[0] = tmp0.u[0]; dst[1] = tmp0.u[1];
            dst[2] = tmp1.u[0]; dst[3] = tmp1.u[1];
        }
        // prefetch next k-tile while this one is consumed
        if (kt + 1 < D_ / 64) {
            if constexpr (MODE == 2)
                __builtin_prefetch(Xh + (size_t)(blockM + lrow) * D_ + k0 + 64 + lcol, 0, 0);
            else
                __builtin_prefetch(Xf + (size_t)(blockM + lrow) * D_ + k0 + 64 + lcol, 0, 0);
            __builtin_prefetch(W + (size_t)(blockN + lrow) * D_ + k0 + 64 + lcol, 0, 0);
        }
        __syncthreads();

#pragma unroll
        for (int kk = 0; kk < 2; ++kk) {
            // preload all fragments for this k-half, then issue WMMAs back-to-back
            Frag16 af[4], bf[2];
#pragma unroll
            for (int i = 0; i < 4; ++i) {
                const uint4* p = reinterpret_cast<const uint4*>(&As[wrow + i * 16 + lm][0]);
                af[i].u[0] = p[kk * 4 + half];        // K = kk*32 + half*8 .. +7
                af[i].u[1] = p[kk * 4 + 2 + half];    // K = kk*32 + 16 + half*8 .. +7
            }
#pragma unroll
            for (int j = 0; j < 2; ++j) {
                const uint4* p = reinterpret_cast<const uint4*>(&Bs[wcol + j * 16 + lm][0]);
                bf[j].u[0] = p[kk * 4 + half];
                bf[j].u[1] = p[kk * 4 + 2 + half];
            }
#pragma unroll
            for (int i = 0; i < 4; ++i)
#pragma unroll
                for (int j = 0; j < 2; ++j)
                    acc[i][j] = __builtin_amdgcn_wmma_f32_16x16x32_f16(
                        false, af[i].h, false, bf[j].h, (short)0, acc[i][j], false, false);
        }
        __syncthreads();
    }

    // ---- epilogue: bias + store ----
#pragma unroll
    for (int i = 0; i < 4; ++i) {
#pragma unroll
        for (int j = 0; j < 2; ++j) {
            const int n  = blockN + wcol + j * 16 + lm;
            const float bn = bias[n];
#pragma unroll
            for (int r = 0; r < 8; ++r) {
                const int m   = blockM + wrow + i * 16 + half * 8 + r;
                const float val = acc[i][j][r] + bn;
                const int bb = m >> 11, s = m & (S_ - 1);
                const int hh = n >> 6,  d = n & (DK_ - 1);
                if constexpr (MODE == 0) {
                    _Float16* out16 = (_Float16*)outp;   // [B,H,S,DK]
                    out16[(((size_t)(bb * H_ + hh) * S_) + s) * DK_ + d] = (_Float16)val;
                } else if constexpr (MODE == 1) {
                    _Float16* out16 = (_Float16*)outp;   // [B*H, DK, S]  (V^T)
                    out16[((size_t)(bb * H_ + hh) * DK_ + d) * S_ + s] = (_Float16)val;
                } else {
                    float* out32 = (float*)outp;         // [M,N] fp32
                    out32[(size_t)m * D_ + n] = val;
                }
            }
        }
    }
}

// ---------------------------------------------------------------------------
// Causal flash attention. One block per (batch*head, 64-query tile).
// 128 threads = 4 waves; each wave owns 16 query rows. DK=64.
// K and V^T tiles are double-buffered in LDS and filled with
// GLOBAL_LOAD_ASYNC_TO_LDS_B128 (ASYNCcnt), pipelined one block ahead.
// ---------------------------------------------------------------------------
__global__ __launch_bounds__(128)
void attn_kernel(const _Float16* __restrict__ Qh, const _Float16* __restrict__ Kh,
                 const _Float16* __restrict__ Vt_g, _Float16* __restrict__ Oh)
{
    __shared__ __align__(16) _Float16 Ks[2][64][72];   // [buf][key][d]
    __shared__ __align__(16) _Float16 Vt[2][64][72];   // [buf][d][key]
    __shared__ __align__(16) _Float16 Ps[64][72];      // [query][key]

    const int t    = threadIdx.x;
    const int wv   = t >> 5;
    const int lane = t & 31;
    const int half = lane >> 4;
    const int lm   = lane & 15;

    const int qtile = blockIdx.x;      // 0..31
    const int bh    = blockIdx.y;      // 0..31
    const int bb    = bh >> 4;
    const int hh    = bh & (H_ - 1);

    const size_t headOff = (size_t)bh * S_ * DK_;

    const int ldr = t >> 1;            // 0..63
    const int ldc = (t & 1) * 32;      // 0/32 (halves) -> 64B segment

    // Issue async copies of K tile [key][d] and V^T tile [d][key] for block kb.
    auto issue_tile = [&](int kb, int buf) {
        const unsigned long long kga = (unsigned long long)(uintptr_t)(
            Kh + headOff + (size_t)(kb * 64 + ldr) * DK_ + ldc);
        const unsigned int klds = (unsigned int)(uintptr_t)&Ks[buf][ldr][ldc];
        asm volatile(
            "global_load_async_to_lds_b128 %0, %1, off\n\t"
            "global_load_async_to_lds_b128 %0, %1, off offset:16\n\t"
            "global_load_async_to_lds_b128 %0, %1, off offset:32\n\t"
            "global_load_async_to_lds_b128 %0, %1, off offset:48"
            :: "v"(klds), "v"(kga) : "memory");
        const unsigned long long vga = (unsigned long long)(uintptr_t)(
            Vt_g + ((size_t)bh * DK_ + ldr) * S_ + kb * 64 + ldc);
        const unsigned int vlds = (unsigned int)(uintptr_t)&Vt[buf][ldr][ldc];
        asm volatile(
            "global_load_async_to_lds_b128 %0, %1, off\n\t"
            "global_load_async_to_lds_b128 %0, %1, off offset:16\n\t"
            "global_load_async_to_lds_b128 %0, %1, off offset:32\n\t"
            "global_load_async_to_lds_b128 %0, %1, off offset:48"
            :: "v"(vlds), "v"(vga) : "memory");
    };

    // Preload this wave's Q fragments (16 rows x 64 d) once.
    Frag16 qa[2];
    {
        const int q0 = qtile * 64 + wv * 16 + lm;
        const uint4* qp = reinterpret_cast<const uint4*>(Qh + headOff + (size_t)q0 * DK_);
        qa[0].u[0] = qp[half];     qa[0].u[1] = qp[2 + half];   // d 0..31
        qa[1].u[0] = qp[4 + half]; qa[1].u[1] = qp[6 + half];   // d 32..63
    }

    v8f o[4] = {};
    float mrow[8], lsum[8];
#pragma unroll
    for (int r = 0; r < 8; ++r) { mrow[r] = -1e30f; lsum[r] = 0.f; }

    issue_tile(0, 0);

    for (int kb = 0; kb <= qtile; ++kb) {
        const int cur = kb & 1;
        // tile kb complete for this wave; barrier makes it block-wide visible
        // and guarantees everyone is done reading buffer cur^1.
        asm volatile("s_wait_asynccnt 0x0" ::: "memory");
        __syncthreads();
        if (kb < qtile) issue_tile(kb + 1, cur ^ 1);   // overlap with compute

        // ---- S = Q @ K^T (16x64 per wave): preload all frags, then 8 WMMAs ----
        Frag16 bfr[2][4];
#pragma unroll
        for (int kk = 0; kk < 2; ++kk)
#pragma unroll
            for (int j = 0; j < 4; ++j) {
                const uint4* p = reinterpret_cast<const uint4*>(&Ks[cur][j * 16 + lm][0]);
                bfr[kk][j].u[0] = p[kk * 4 + half];
                bfr[kk][j].u[1] = p[kk * 4 + 2 + half];
            }
        v8f sacc[4] = {};
#pragma unroll
        for (int kk = 0; kk < 2; ++kk)
#pragma unroll
            for (int j = 0; j < 4; ++j)
                sacc[j] = __builtin_amdgcn_wmma_f32_16x16x32_f16(
                    false, qa[kk].h, false, bfr[kk][j].h, (short)0, sacc[j], false, false);

        // ---- online softmax + write P (f16, wave-private rows) ----
#pragma unroll
        for (int r = 0; r < 8; ++r) {
            const int qg = qtile * 64 + wv * 16 + half * 8 + r;
            float vals[4];
#pragma unroll
            for (int j = 0; j < 4; ++j) {
                const int key = kb * 64 + j * 16 + lm;
                const float x = sacc[j][r] * 0.125f;   // 1/sqrt(64)
                vals[j] = (key > qg) ? -1e30f : x;
            }
            float rmax = fmaxf(fmaxf(vals[0], vals[1]), fmaxf(vals[2], vals[3]));
#pragma unroll
            for (int off = 1; off <= 8; off <<= 1)
                rmax = fmaxf(rmax, __shfl_xor(rmax, off, 32));
            const float mnew  = fmaxf(mrow[r], rmax);
            const float alpha = __expf(mrow[r] - mnew);
            float rsum = 0.f;
#pragma unroll
            for (int j = 0; j < 4; ++j) {
                const float p = __expf(vals[j] - mnew);
                rsum += p;
                Ps[wv * 16 + half * 8 + r][j * 16 + lm] = (_Float16)p;
            }
#pragma unroll
            for (int off = 1; off <= 8; off <<= 1)
                rsum += __shfl_xor(rsum, off, 32);
            lsum[r] = lsum[r] * alpha + rsum;
            mrow[r] = mnew;
#pragma unroll
            for (int tt = 0; tt < 4; ++tt) o[tt][r] *= alpha;
        }
        // Ps rows are produced and consumed by the same wave; same-wave LDS
        // ops are in-order (DScnt), so no block barrier is needed here.

        // ---- O += P @ V: preload all frags, then 8 WMMAs ----
        Frag16 pa[2], vfr[2][4];
        {
            const uint4* pp = reinterpret_cast<const uint4*>(&Ps[wv * 16 + lm][0]);
#pragma unroll
            for (int kk = 0; kk < 2; ++kk) {
                pa[kk].u[0] = pp[kk * 4 + half];
                pa[kk].u[1] = pp[kk * 4 + 2 + half];
            }
        }
#pragma unroll
        for (int kk = 0; kk < 2; ++kk)
#pragma unroll
            for (int tt = 0; tt < 4; ++tt) {
                const uint4* vp2 = reinterpret_cast<const uint4*>(&Vt[cur][tt * 16 + lm][0]);
                vfr[kk][tt].u[0] = vp2[kk * 4 + half];
                vfr[kk][tt].u[1] = vp2[kk * 4 + 2 + half];
            }
#pragma unroll
        for (int kk = 0; kk < 2; ++kk)
#pragma unroll
            for (int tt = 0; tt < 4; ++tt)
                o[tt] = __builtin_amdgcn_wmma_f32_16x16x32_f16(
                    false, pa[kk].h, false, vfr[kk][tt].h, (short)0, o[tt], false, false);
    }

    // ---- normalize + write O as f16 [B,S,D] (heads concatenated) ----
#pragma unroll
    for (int tt = 0; tt < 4; ++tt) {
#pragma unroll
        for (int r = 0; r < 8; ++r) {
            const int srow = qtile * 64 + wv * 16 + half * 8 + r;
            const int col  = hh * DK_ + tt * 16 + lm;
            const float ov = o[tt][r] / lsum[r];
            Oh[((size_t)bb * S_ + srow) * D_ + col] = (_Float16)ov;
        }
    }
}

// ---------------------------------------------------------------------------
// inputs: q,k,v,mask,wq,bq,wk,bk,wv,bv,wo,bo  (mask is causal tril -> applied
// analytically in attn_kernel; the mask buffer is unused)
// ---------------------------------------------------------------------------
extern "C" void kernel_launch(void* const* d_in, const int* in_sizes, int n_in,
                              void* d_out, int out_size, void* d_ws, size_t ws_size,
                              hipStream_t stream)
{
    (void)in_sizes; (void)n_in; (void)out_size; (void)ws_size;

    const float* q  = (const float*)d_in[0];
    const float* k  = (const float*)d_in[1];
    const float* v  = (const float*)d_in[2];
    const float* wq = (const float*)d_in[4];
    const float* bq = (const float*)d_in[5];
    const float* wk = (const float*)d_in[6];
    const float* bk = (const float*)d_in[7];
    const float* wv = (const float*)d_in[8];
    const float* bv = (const float*)d_in[9];
    const float* wo = (const float*)d_in[10];
    const float* bo = (const float*)d_in[11];

    const size_t elems = (size_t)B_ * S_ * D_;   // 4,194,304
    _Float16* Qh  = (_Float16*)d_ws;
    _Float16* Kh  = Qh + elems;
    _Float16* Vth = Kh + elems;                  // V^T per head: [B*H, DK, S]
    _Float16* Oh  = Vth + elems;

    dim3 gemmGrid(D_ / 128, (B_ * S_) / 128);    // (8, 32)
    gemm_xwT_kernel<0><<<gemmGrid, 256, 0, stream>>>(q, wq, bq, Qh);
    gemm_xwT_kernel<0><<<gemmGrid, 256, 0, stream>>>(k, wk, bk, Kh);
    gemm_xwT_kernel<1><<<gemmGrid, 256, 0, stream>>>(v, wv, bv, Vth);

    attn_kernel<<<dim3(S_ / 64, B_ * H_), 128, 0, stream>>>(Qh, Kh, Vth, Oh);

    gemm_xwT_kernel<2><<<gemmGrid, 256, 0, stream>>>(Oh, wo, bo, (float*)d_out);
}
